// SparseMoE_56392920597058
// MI455X (gfx1250) — matrix operand order repaired and checked by reading, not compiled, not run on hardware
//
#include <hip/hip_runtime.h>

#define TOKENS 8192
#define DM 1024
#define DH 4096
#define NE 8

typedef __bf16 v16bf __attribute__((ext_vector_type(16)));
typedef float  v8f   __attribute__((ext_vector_type(8)));

__device__ __forceinline__ unsigned short f2bf(float f) {
  unsigned u = __float_as_uint(f);
  u += 0x7fffu + ((u >> 16) & 1u);   // round-to-nearest-even bf16
  return (unsigned short)(u >> 16);
}

union FragBF { v16bf v; uint4 q[2]; };

__device__ __forceinline__ v16bf ldsFrag(const unsigned* lds, int u0, int u1) {
  FragBF f;
  f.q[0] = *(const uint4*)(lds + u0);
  f.q[1] = *(const uint4*)(lds + u1);
  return f.v;
}

// LDS byte offset of a __shared__ object (as3->flat cast keeps offset in low 32 bits)
__device__ __forceinline__ unsigned ldsOff(const void* p) {
  return (unsigned)(unsigned long long)p;
}

// async global->LDS 16B copy, tracked by ASYNCcnt (GV addressing mode)
__device__ __forceinline__ void asyncLd128(unsigned lds, const void* g) {
  asm volatile("global_load_async_to_lds_b128 %0, %1, off"
               :: "v"(lds), "v"((unsigned long long)g) : "memory");
}

__device__ __forceinline__ void waitAsync0() {
#if defined(__has_builtin)
#if __has_builtin(__builtin_amdgcn_s_wait_asynccnt)
  __builtin_amdgcn_s_wait_asynccnt(0);
  return;
#endif
#endif
  asm volatile("s_wait_asynccnt 0x0" ::: "memory");
}

// ---------------- zero out + counters ----------------
__global__ void moe_zero(float* out, int* cnt) {
  int i = blockIdx.x * blockDim.x + threadIdx.x;
  if (i < NE) cnt[i] = 0;
  for (int j = i; j < TOKENS * DM; j += gridDim.x * blockDim.x) out[j] = 0.f;
}

// ---------------- weight fp32 [e][K][N] -> bf16 transposed [e][N][K] ----------------
__global__ __launch_bounds__(256) void moe_cvtw(const float* __restrict__ W,
                                                unsigned short* __restrict__ Wt,
                                                int K, int N) {
  __shared__ unsigned short t[32][33];
  const int e = blockIdx.z;
  const float* src = W + (size_t)e * K * N;
  unsigned short* dst = Wt + (size_t)e * N * K;
  const int n0 = blockIdx.x * 32, k0 = blockIdx.y * 32;
  const int c = threadIdx.x & 31, r8 = threadIdx.x >> 5;
#pragma unroll
  for (int rr = 0; rr < 4; ++rr) {
    int k = r8 + rr * 8;
    t[k][c] = f2bf(src[(size_t)(k0 + k) * N + n0 + c]);
  }
  __syncthreads();
#pragma unroll
  for (int rr = 0; rr < 4; ++rr) {
    int n = r8 + rr * 8;
    dst[(size_t)(n0 + n) * K + k0 + c] = t[c][n];
  }
}

// ---------------- gating: logits -> top-2 -> expert lists ----------------
__global__ __launch_bounds__(256) void moe_gate(const float* __restrict__ x0,
                                                const float* __restrict__ Wg,
                                                int* __restrict__ cnt,
                                                int* __restrict__ idx) {
  __shared__ float sWg[DM * NE];   // 32 KB
  const int tid = threadIdx.x;
  for (int i = tid; i < DM * NE; i += 256) sWg[i] = Wg[i];
  __syncthreads();

  const int t = blockIdx.x * 256 + tid;
  float acc[NE];
#pragma unroll
  for (int e = 0; e < NE; ++e) acc[e] = 0.f;
  const float4* xr = (const float4*)(x0 + (size_t)t * DM);
  for (int d4 = 0; d4 < DM / 4; ++d4) {
    float4 x = xr[d4];
    const float* w = &sWg[d4 * 4 * NE];
#pragma unroll
    for (int e = 0; e < NE; ++e)
      acc[e] += x.x * w[e] + x.y * w[NE + e] + x.z * w[2 * NE + e] + x.w * w[3 * NE + e];
  }
  // top-2 of logits (softmax monotonic; combine is unweighted); '>' matches jax tie-break
  int i1 = 0;
#pragma unroll
  for (int e = 1; e < NE; ++e) if (acc[e] > acc[i1]) i1 = e;
  int i2 = (i1 == 0) ? 1 : 0;
#pragma unroll
  for (int e = 0; e < NE; ++e) if (e != i1 && acc[e] > acc[i2]) i2 = e;
  int s1 = atomicAdd(&cnt[i1], 1); idx[i1 * TOKENS + s1] = t;
  int s2 = atomicAdd(&cnt[i2], 1); idx[i2 * TOKENS + s2] = t;
}

// ---------------- tiny exclusive scan for compact H row bases ----------------
__global__ void moe_scan(const int* __restrict__ cnt, int* __restrict__ base) {
  if (threadIdx.x == 0) {
    int s = 0;
    for (int e = 0; e < NE; ++e) { base[e] = s; s += cnt[e]; }
  }
}

// ---------------- FFN layer 1: H = relu(gather(xl) @ W1[e] + b1[e]) ----------------
// Block tile 128x256, 8 waves as 2x4, each wave 64x64 (4x4 WMMA accs).
// A: gathered fp32 rows -> cvt -> LDS (pipelined through registers)
// B: bf16 pre-transposed weights -> async copy to LDS (double buffered)
__global__ __launch_bounds__(256) void moe_ffn1(
    const float* __restrict__ xl, const unsigned short* __restrict__ W1t,
    const float* __restrict__ b1, const int* __restrict__ cnt,
    const int* __restrict__ base, const int* __restrict__ idx,
    unsigned short* __restrict__ H) {
  const int e  = blockIdx.z;
  const int cn = cnt[e];
  const int mt = blockIdx.x;
  if (mt * 128 >= cn) return;
  const int n0 = blockIdx.y * 256;
  const int hb = base[e];

  __shared__ unsigned sA[2][128 * 20];   // 128 rows x 32 bf16, 80B padded stride
  __shared__ unsigned sB[2][256 * 20];   // 256 cols x 32 bf16 (k contiguous), 80B stride

  const int tid = threadIdx.x;
  // A loader: 2 threads per row, 16 consecutive k each (gathered fp32 row)
  const int arow = tid >> 1, akseg = (tid & 1) * 16;
  const int aslot = mt * 128 + arow;
  const int atok = idx[e * TOKENS + (aslot < cn ? aslot : 0)];
  const float* aptr = xl + (size_t)atok * DM + akseg;
  const unsigned aLds = arow * 80 + akseg * 2;   // byte offset within buffer
  // B loader: 1 thread per col, full 32-k column (64B = 4x b128 async)
  const int bcol = tid;
  const unsigned short* bptr = W1t + ((size_t)e * DH + n0 + bcol) * DM;
  const unsigned bLds = bcol * 80;
  const unsigned sBo[2] = { ldsOff(&sB[0][0]), ldsOff(&sB[1][0]) };

  const int lane = tid & 31, wave = tid >> 5;
  const int wm = wave >> 2, wn = wave & 3;            // 2x4 wave grid -> 64x64 per wave
  const int laneN = lane & 15, laneHalf = lane >> 4;

  v8f zero = {};
  v8f acc[4][4];
#pragma unroll
  for (int i = 0; i < 4; ++i)
#pragma unroll
    for (int j = 0; j < 4; ++j) acc[i][j] = zero;

  // ---- prologue: stage 0 into buf0, prefetch stage 1 A into registers ----
  float4 aReg[4];
#pragma unroll
  for (int q = 0; q < 4; ++q) aReg[q] = ((const float4*)aptr)[q];
  {
    unsigned u[8];
#pragma unroll
    for (int q = 0; q < 4; ++q) {
      float4 f = aReg[q];
      u[q * 2]     = (unsigned)f2bf(f.x) | ((unsigned)f2bf(f.y) << 16);
      u[q * 2 + 1] = (unsigned)f2bf(f.z) | ((unsigned)f2bf(f.w) << 16);
    }
    uint4* d = (uint4*)((char*)&sA[0][0] + aLds);
    d[0] = make_uint4(u[0], u[1], u[2], u[3]);
    d[1] = make_uint4(u[4], u[5], u[6], u[7]);
  }
#pragma unroll
  for (int q = 0; q < 4; ++q) asyncLd128(sBo[0] + bLds + q * 16, bptr + q * 8);
#pragma unroll
  for (int q = 0; q < 4; ++q) aReg[q] = ((const float4*)(aptr + 32))[q];

  for (int kt = 0, it = 0; kt < DM; kt += 32, ++it) {
    const int buf = it & 1;
    waitAsync0();
    __syncthreads();
    if (kt + 32 < DM) {
      const int nb = buf ^ 1;
#pragma unroll
      for (int q = 0; q < 4; ++q)
        asyncLd128(sBo[nb] + bLds + q * 16, bptr + kt + 32 + q * 8);
      unsigned u[8];
#pragma unroll
      for (int q = 0; q < 4; ++q) {
        float4 f = aReg[q];
        u[q * 2]     = (unsigned)f2bf(f.x) | ((unsigned)f2bf(f.y) << 16);
        u[q * 2 + 1] = (unsigned)f2bf(f.z) | ((unsigned)f2bf(f.w) << 16);
      }
      uint4* d = (uint4*)((char*)&sA[nb][0] + aLds);
      d[0] = make_uint4(u[0], u[1], u[2], u[3]);
      d[1] = make_uint4(u[4], u[5], u[6], u[7]);
      if (kt + 64 < DM) {
#pragma unroll
        for (int q = 0; q < 4; ++q) aReg[q] = ((const float4*)(aptr + kt + 64))[q];
      }
    }
    const unsigned* cA = sA[buf];
    const unsigned* cB = sB[buf];
    v16bf af[4], bfr[4];
#pragma unroll
    for (int i = 0; i < 4; ++i) {   // A frag: lanes0-15 K0-7|K16-23, lanes16-31 K8-15|K24-31
      int r = wm * 64 + i * 16 + laneN;
      af[i] = ldsFrag(cA, r * 20 + laneHalf * 4, r * 20 + 8 + laneHalf * 4);
    }
#pragma unroll
    for (int j = 0; j < 4; ++j) {   // B frag: lanes0-15 K0-7|K16-23, lanes16-31 K8-15|K24-31
      int c = wn * 64 + j * 16 + laneN;
      bfr[j] = ldsFrag(cB, c * 20 + laneHalf * 4, c * 20 + 8 + laneHalf * 4);
    }
#pragma unroll
    for (int i = 0; i < 4; ++i)
#pragma unroll
      for (int j = 0; j < 4; ++j)
        acc[i][j] = __builtin_amdgcn_wmma_f32_16x16x32_bf16(
            false, af[i], false, bfr[j], (short)0, acc[i][j], false, false);
  }
  // epilogue: +b1, relu, bf16 store into compact H
#pragma unroll
  for (int j = 0; j < 4; ++j) {
    const int n = n0 + wn * 64 + j * 16 + laneN;
    const float bias = b1[e * DH + n];
#pragma unroll
    for (int i = 0; i < 4; ++i) {
      const int slotBase = mt * 128 + wm * 64 + i * 16 + laneHalf * 8;
#pragma unroll
      for (int v = 0; v < 8; ++v) {
        const int slot = slotBase + v;
        if (slot < cn) {
          float h = fmaxf(acc[i][j][v] + bias, 0.f);
          H[(size_t)(hb + slot) * DH + n] = f2bf(h);
        }
      }
    }
  }
}

// ---------------- FFN layer 2: out[token] += H @ W2[e] + b2[e] ----------------
// A (H) and B (bf16 transposed W2) both async-copied to LDS, double buffered
__global__ __launch_bounds__(256) void moe_ffn2(
    const unsigned short* __restrict__ H, const unsigned short* __restrict__ W2t,
    const float* __restrict__ b2, const int* __restrict__ cnt,
    const int* __restrict__ base, const int* __restrict__ idx,
    float* __restrict__ out) {
  const int e  = blockIdx.z;
  const int cn = cnt[e];
  const int mt = blockIdx.x;
  if (mt * 128 >= cn) return;
  const int n0 = blockIdx.y * 256;
  const int hb = base[e];

  __shared__ unsigned sA[2][128 * 20];
  __shared__ unsigned sB[2][256 * 20];

  const int tid = threadIdx.x;
  const int arow = tid >> 1, akseg = (tid & 1) * 16;
  const int aslot = mt * 128 + arow;
  const int hrow = hb + (aslot < cn ? aslot : 0);
  const unsigned short* aptr = H + (size_t)hrow * DH + akseg;
  const unsigned aLds = arow * 80 + akseg * 2;
  const int bcol = tid;
  const unsigned short* bptr = W2t + ((size_t)e * DM + n0 + bcol) * DH;
  const unsigned bLds = bcol * 80;
  const unsigned sAo[2] = { ldsOff(&sA[0][0]), ldsOff(&sA[1][0]) };
  const unsigned sBo[2] = { ldsOff(&sB[0][0]), ldsOff(&sB[1][0]) };

  const int lane = tid & 31, wave = tid >> 5;
  const int wm = wave >> 2, wn = wave & 3;
  const int laneN = lane & 15, laneHalf = lane >> 4;

  v8f zero = {};
  v8f acc[4][4];
#pragma unroll
  for (int i = 0; i < 4; ++i)
#pragma unroll
    for (int j = 0; j < 4; ++j) acc[i][j] = zero;

  // prologue: stage 0 into buf0
  asyncLd128(sAo[0] + aLds,      aptr);
  asyncLd128(sAo[0] + aLds + 16, aptr + 8);
#pragma unroll
  for (int q = 0; q < 4; ++q) asyncLd128(sBo[0] + bLds + q * 16, bptr + q * 8);

  for (int kt = 0, it = 0; kt < DH; kt += 32, ++it) {
    const int buf = it & 1;
    waitAsync0();
    __syncthreads();
    if (kt + 32 < DH) {
      const int nb = buf ^ 1;
      asyncLd128(sAo[nb] + aLds,      aptr + kt + 32);
      asyncLd128(sAo[nb] + aLds + 16, aptr + kt + 40);
#pragma unroll
      for (int q = 0; q < 4; ++q)
        asyncLd128(sBo[nb] + bLds + q * 16, bptr + kt + 32 + q * 8);
    }
    const unsigned* cA = sA[buf];
    const unsigned* cB = sB[buf];
    v16bf af[4], bfr[4];
#pragma unroll
    for (int i = 0; i < 4; ++i) {
      int r = wm * 64 + i * 16 + laneN;
      af[i] = ldsFrag(cA, r * 20 + laneHalf * 4, r * 20 + 8 + laneHalf * 4);
    }
#pragma unroll
    for (int j = 0; j < 4; ++j) {
      int c = wn * 64 + j * 16 + laneN;
      bfr[j] = ldsFrag(cB, c * 20 + laneHalf * 4, c * 20 + 8 + laneHalf * 4);
    }
#pragma unroll
    for (int i = 0; i < 4; ++i)
#pragma unroll
      for (int j = 0; j < 4; ++j)
        acc[i][j] = __builtin_amdgcn_wmma_f32_16x16x32_bf16(
            false, af[i], false, bfr[j], (short)0, acc[i][j], false, false);
  }
  // epilogue: +b2, scatter-add into out[token]
#pragma unroll
  for (int i = 0; i < 4; ++i) {
    const int slotBase = mt * 128 + wm * 64 + i * 16 + laneHalf * 8;
    int tok[8];
#pragma unroll
    for (int v = 0; v < 8; ++v) {
      const int slot = slotBase + v;
      tok[v] = (slot < cn) ? idx[e * TOKENS + slot] : -1;
    }
#pragma unroll
    for (int j = 0; j < 4; ++j) {
      const int n = n0 + wn * 64 + j * 16 + laneN;
      const float bias = b2[e * DM + n];
#pragma unroll
      for (int v = 0; v < 8; ++v)
        if (tok[v] >= 0)
          unsafeAtomicAdd(out + (size_t)tok[v] * DM + n, acc[i][j][v] + bias);
    }
  }
}

extern "C" void kernel_launch(void* const* d_in, const int* in_sizes, int n_in,
                              void* d_out, int out_size, void* d_ws, size_t ws_size,
                              hipStream_t stream) {
  (void)in_sizes; (void)n_in; (void)out_size; (void)ws_size;
  const float* xl = (const float*)d_in[0];
  const float* x0 = (const float*)d_in[1];
  const float* Wg = (const float*)d_in[2];
  const float* W1 = (const float*)d_in[3];
  const float* b1 = (const float*)d_in[4];
  const float* W2 = (const float*)d_in[5];
  const float* b2 = (const float*)d_in[6];
  float* out = (float*)d_out;

  // ws: [cnt 8i][base 8i][pad->256B][idx 8*8192 i][H 16384*4096 bf16]
  //     [W1t 8*4096*1024 bf16][W2t 8*1024*4096 bf16]   total ~256.3 MiB
  char* ws = (char*)d_ws;
  int* cnt  = (int*)ws;
  int* base = cnt + NE;
  int* idx  = (int*)(ws + 256);
  unsigned short* H   = (unsigned short*)(ws + 256 + sizeof(int) * NE * TOKENS);
  unsigned short* W1t = (unsigned short*)((char*)H + (size_t)2 * TOKENS * DH * 2);
  unsigned short* W2t = (unsigned short*)((char*)W1t + (size_t)NE * DM * DH * 2);

  moe_zero<<<4096, 256, 0, stream>>>(out, cnt);
  moe_cvtw<<<dim3(DH / 32, DM / 32, NE), 256, 0, stream>>>(W1, W1t, DM, DH);
  moe_cvtw<<<dim3(DM / 32, DH / 32, NE), 256, 0, stream>>>(W2, W2t, DH, DM);
  moe_gate<<<TOKENS / 256, 256, 0, stream>>>(x0, Wg, cnt, idx);
  moe_scan<<<1, 32, 0, stream>>>(cnt, base);
  moe_ffn1<<<dim3(64, DH / 256, NE), 256, 0, stream>>>(xl, W1t, b1, cnt, base, idx, H);
  moe_ffn2<<<dim3(64, DM / 256, NE), 256, 0, stream>>>(H, W2t, b2, cnt, base, idx, out);
}
